// CyCTransformer_72739566125133
// MI455X (gfx1250) — matrix operand order repaired and checked by reading, not compiled, not run on hardware
//
#include <hip/hip_runtime.h>

// ---------------- constants (fixed by the reference) ----------------
#define CDIM 384
#define NTOK 4096        // 64*64
#define NHEAD 12
#define DHEAD 32
#define NPTS 9
#define NLAYER 2
#define NN 256           // 16*16 after stride-4 SR conv
#define KCONV 6144       // CDIM*4*4

typedef __attribute__((ext_vector_type(16))) _Float16 v16h;
typedef __attribute__((ext_vector_type(4)))  _Float16 v4h;
typedef __attribute__((ext_vector_type(8)))  float    v8f;

// LDS: fragment-order tiles, one 16-wide block per 32 lanes, 16 halfs used,
// padded to 24 (48 B, 16 B-aligned) per lane to spread banks.
#define FRAG_STRIDE 24

// ---------------- wave32 reductions ----------------
__device__ inline float wave_sum(float v){
#pragma unroll
  for (int o = 16; o > 0; o >>= 1) v += __shfl_xor(v, o, 32);
  return v;
}
__device__ inline float wave_max(float v){
#pragma unroll
  for (int o = 16; o > 0; o >>= 1) v = fmaxf(v, __shfl_xor(v, o, 32));
  return v;
}

// ---------------- batched WMMA GEMM ----------------
// C[z] = alpha * A[z](MxK) * B[z](KxN) + bias ; optional ReLU; TRANSB template.
// Preconditions (hold for every call in this net): M % 128 == 0, K % 32 == 0.
// Two-level batch: z = outer*innerB + inner with independent strides (for (b,head)).
//
// LDS swizzle (this code's contiguous k order, same for A and B; not the ISA table order):
//  A 16x32 f16 frag: lane = (k/16)*16 + m, component c = k%16
//  B 32x16 f16 frag: lane = (k/16)*16 + n%16, component c = k%16
// Tiles are staged already in fragment order -> frag load = one contiguous v16h.
template<int TRANSB>
__global__ __launch_bounds__(256)
void gemm_wmma_f16(const float* __restrict__ A, const float* __restrict__ Bm,
                   const float* __restrict__ bias, float* __restrict__ C,
                   int M, int N, int K,
                   long lda, long ldb, long ldc,
                   long sAo, long sAi, long sBo, long sBi, long sCo, long sCi,
                   int innerB, float alpha, int relu)
{
  __shared__ _Float16 As[8 * 32 * FRAG_STRIDE];   // 128(M) x 32(K)
  __shared__ _Float16 Bs[8 * 32 * FRAG_STRIDE];   // 32(K) x 128(N)

  int z = blockIdx.z;
  int outer = z / innerB, inner = z % innerB;
  A  += (long)outer * sAo + (long)inner * sAi;
  Bm += (long)outer * sBo + (long)inner * sBi;
  C  += (long)outer * sCo + (long)inner * sCi;

  int m0 = blockIdx.y * 128;
  int n0 = blockIdx.x * 128;
  int tid = threadIdx.x;
  int lane = tid & 31, wave = tid >> 5;
  int wm = wave >> 1, wn = wave & 1;     // 4x2 wave grid; each wave: 32(M) x 64(N)

  v8f zero = {};
  v8f acc[2][4];
#pragma unroll
  for (int mi = 0; mi < 2; ++mi)
#pragma unroll
    for (int ni = 0; ni < 4; ++ni) acc[mi][ni] = zero;

  for (int k0 = 0; k0 < K; k0 += 32){
    bool pf = (k0 + 32) < K;
    // ---- stage A tile (128x32): 4 float4 per thread, no bounds checks ----
#pragma unroll
    for (int t = 0; t < 4; ++t){
      int idx = t * 256 + tid;
      int r = idx >> 3, kq = (idx & 7) * 4;
      const float* ap = A + (long)(m0 + r) * lda + (k0 + kq);
      float4 f = *(const float4*)ap;
      if (pf) __builtin_prefetch(ap + 32, 0, 1);
      int g  = kq >> 4;
      int c0 = kq & 15;
      int off = (((r >> 4) * 32) + g * 16 + (r & 15)) * FRAG_STRIDE + c0;
      v4h h; h[0] = (_Float16)f.x; h[1] = (_Float16)f.y;
             h[2] = (_Float16)f.z; h[3] = (_Float16)f.w;
      *(v4h*)(As + off) = h;
    }
    // ---- stage B tile (32x128): branchless N-edge (clamp + select) ----
    if (TRANSB){
#pragma unroll
      for (int t = 0; t < 4; ++t){
        int idx = t * 256 + tid;
        int n = idx >> 3, kq = (idx & 7) * 4;
        int gn = n0 + n;
        int gns = gn < N ? gn : (N - 1);
        const float* bp = Bm + (long)gns * ldb + (k0 + kq);
        float4 f = *(const float4*)bp;
        if (pf) __builtin_prefetch(bp + 32, 0, 1);
        bool ok = gn < N;
        f.x = ok ? f.x : 0.f; f.y = ok ? f.y : 0.f;
        f.z = ok ? f.z : 0.f; f.w = ok ? f.w : 0.f;
        int off = ((n >> 4) * 32 + (kq >> 4) * 16 + (n & 15)) * FRAG_STRIDE + (kq & 15);
        v4h h; h[0] = (_Float16)f.x; h[1] = (_Float16)f.y;
               h[2] = (_Float16)f.z; h[3] = (_Float16)f.w;
        *(v4h*)(Bs + off) = h;
      }
    } else {
#pragma unroll
      for (int t = 0; t < 4; ++t){
        int idx = t * 256 + tid;
        int k = idx >> 5, nq = (idx & 31) * 4;
        int gk = k0 + k;
        int gn0 = n0 + nq;
        const float* bp = Bm + (long)gk * ldb + gn0;
        float4 f;
        if (gn0 + 3 < N){
          f = *(const float4*)bp;
        } else {
          float vv[4];
#pragma unroll
          for (int e = 0; e < 4; ++e){
            int gn = gn0 + e;
            int gns = gn < N ? gn : (N - 1);
            float v = Bm[(long)gk * ldb + gns];
            vv[e] = gn < N ? v : 0.f;
          }
          f.x = vv[0]; f.y = vv[1]; f.z = vv[2]; f.w = vv[3];
        }
        if (pf) __builtin_prefetch(Bm + (long)(gk + 32) * ldb + gn0, 0, 1);
        int c = k & 15, gkk = k >> 4;
        float fr[4] = {f.x, f.y, f.z, f.w};
#pragma unroll
        for (int e = 0; e < 4; ++e){
          int n = nq + e;
          Bs[((n >> 4) * 32 + gkk * 16 + (n & 15)) * FRAG_STRIDE + c] = (_Float16)fr[e];
        }
      }
    }
    __syncthreads();

    // ---- fragments: contiguous 32 B per lane (2x ds_load_b128, no repack) ----
    v16h afr[2], bfr[4];
#pragma unroll
    for (int mi = 0; mi < 2; ++mi)
      afr[mi] = *(const v16h*)(As + ((wm * 2 + mi) * 32 + lane) * FRAG_STRIDE);
#pragma unroll
    for (int ni = 0; ni < 4; ++ni)
      bfr[ni] = *(const v16h*)(Bs + ((wn * 4 + ni) * 32 + lane) * FRAG_STRIDE);

#pragma unroll
    for (int mi = 0; mi < 2; ++mi)
#pragma unroll
      for (int ni = 0; ni < 4; ++ni)
        acc[mi][ni] = __builtin_amdgcn_wmma_f32_16x16x32_f16(
            false, afr[mi], false, bfr[ni], (short)0, acc[mi][ni], false, false);
    __syncthreads();
  }

  // C/D layout: VGPR c -> M = c + 8*(lane/16), N = lane%16.  M edge impossible.
  int nloc = lane & 15, g = lane >> 4;
#pragma unroll
  for (int mi = 0; mi < 2; ++mi)
#pragma unroll
    for (int ni = 0; ni < 4; ++ni)
#pragma unroll
      for (int c = 0; c < 8; ++c){
        int row = m0 + wm * 32 + mi * 16 + c + 8 * g;
        int col = n0 + wn * 64 + ni * 16 + nloc;
        if (col < N){
          float v = acc[mi][ni][c] * alpha + (bias ? bias[col] : 0.f);
          if (relu) v = fmaxf(v, 0.f);
          C[(long)row * ldc + col] = v;
        }
      }
}

// ---------------- elementwise / normalization kernels ----------------
__global__ void k_fill_zero(float* p, long n){
  long i = (long)blockIdx.x * 256 + threadIdx.x;
  if (i < n) p[i] = 0.f;
}

__global__ void k_transpose_x(const float* __restrict__ x, float* __restrict__ q, int B){
  long total = (long)B * NTOK * CDIM;
  long i = (long)blockIdx.x * 256 + threadIdx.x;
  if (i >= total) return;
  int c = (int)(i % CDIM); long r = i / CDIM;
  int n = (int)(r % NTOK); int b = (int)(r / NTOK);
  q[i] = x[((long)b * CDIM + c) * NTOK + n];
}

// sine positional encoding + level embed
__global__ void k_pos(const float* __restrict__ lvl, float* __restrict__ pos){
  long i = (long)blockIdx.x * 256 + threadIdx.x;
  if (i >= (long)NTOK * CDIM) return;
  int c = (int)(i % CDIM); int n = (int)(i / CDIM);
  int y = n >> 6, x = n & 63;
  const int nf = CDIM / 2;
  const float scale = 6.283185307179586f;
  float e; int d;
  if (c < nf){ e = (y + 1.0f) / (64.0f + 1e-6f) * scale; d = c; }
  else       { e = (x + 1.0f) / (64.0f + 1e-6f) * scale; d = c - nf; }
  float dt = powf(10000.0f, 2.0f * (float)(d >> 1) / (float)nf);
  float t = e / dt;
  pos[i] = (((d & 1) == 0) ? sinf(t) : cosf(t)) + lvl[c];
}

__global__ void k_add_pos(const float* __restrict__ q, const float* __restrict__ pos,
                          float* __restrict__ out, int B){
  long total = (long)B * NTOK * CDIM;
  long i = (long)blockIdx.x * 256 + threadIdx.x;
  if (i >= total) return;
  long nc = i % ((long)NTOK * CDIM);
  out[i] = q[i] + pos[nc];
}

// im2col for 4x4 stride-4 conv reading [B,N,C]-layout activations
__global__ void k_im2col(const float* __restrict__ src, long bStride,
                         float* __restrict__ dst, int B){
  long total = (long)B * NN * KCONV;
  long i = (long)blockIdx.x * 256 + threadIdx.x;
  if (i >= total) return;
  int col = (int)(i % KCONV); long row = i / KCONV;
  int pp = (int)(row % NN); int b = (int)(row / NN);
  int py = pp >> 4, px = pp & 15;
  int ic = col >> 4, kh = (col >> 2) & 3, kw = col & 3;
  int n = (py * 4 + kh) * 64 + (px * 4 + kw);
  dst[i] = src[(long)b * bStride + (long)n * CDIM + ic];
}

// LayerNorm over C=384: one wave per row, 12 elements per lane
__global__ void k_layernorm(float* __restrict__ x, const float* __restrict__ g,
                            const float* __restrict__ b, long rows){
  long wid = ((long)blockIdx.x * 256 + threadIdx.x) >> 5;
  int lane = threadIdx.x & 31;
  if (wid >= rows) return;
  float* row = x + wid * CDIM;
  float v[12]; float s = 0.f;
#pragma unroll
  for (int j = 0; j < 12; ++j){ v[j] = row[lane + 32 * j]; s += v[j]; }
  s = wave_sum(s);
  float m = s * (1.0f / CDIM);
  float s2 = 0.f;
#pragma unroll
  for (int j = 0; j < 12; ++j){ float d = v[j] - m; s2 += d * d; }
  s2 = wave_sum(s2);
  float inv = rsqrtf(s2 * (1.0f / CDIM) + 1e-5f);
#pragma unroll
  for (int j = 0; j < 12; ++j){ int c = lane + 32 * j; row[c] = (v[j] - m) * inv * g[c] + b[c]; }
}

// dst = LayerNorm(x + alpha*add)
__global__ void k_res_layernorm(float* __restrict__ dst, const float* __restrict__ x,
                                const float* __restrict__ add, float alpha,
                                const float* __restrict__ g, const float* __restrict__ b,
                                long rows){
  long wid = ((long)blockIdx.x * 256 + threadIdx.x) >> 5;
  int lane = threadIdx.x & 31;
  if (wid >= rows) return;
  const float* xr = x + wid * CDIM;
  const float* ar = add + wid * CDIM;
  float* dr = dst + wid * CDIM;
  float v[12]; float s = 0.f;
#pragma unroll
  for (int j = 0; j < 12; ++j){ int c = lane + 32 * j; v[j] = xr[c] + alpha * ar[c]; s += v[j]; }
  s = wave_sum(s);
  float m = s * (1.0f / CDIM);
  float s2 = 0.f;
#pragma unroll
  for (int j = 0; j < 12; ++j){ float d = v[j] - m; s2 += d * d; }
  s2 = wave_sum(s2);
  float inv = rsqrtf(s2 * (1.0f / CDIM) + 1e-5f);
#pragma unroll
  for (int j = 0; j < 12; ++j){ int c = lane + 32 * j; dr[c] = (v[j] - m) * inv * g[c] + b[c]; }
}

// softmax over rows of 256: one wave per row
__global__ void k_softmax256(float* __restrict__ x, long rows){
  long wid = ((long)blockIdx.x * 256 + threadIdx.x) >> 5;
  int lane = threadIdx.x & 31;
  if (wid >= rows) return;
  float* row = x + wid * 256;
  float v[8]; float mx = -1e30f;
#pragma unroll
  for (int j = 0; j < 8; ++j){ v[j] = row[lane + 32 * j]; mx = fmaxf(mx, v[j]); }
  mx = wave_max(mx);
  float s = 0.f;
#pragma unroll
  for (int j = 0; j < 8; ++j){ v[j] = __expf(v[j] - mx); s += v[j]; }
  s = wave_sum(s);
  float inv = 1.0f / s;
#pragma unroll
  for (int j = 0; j < 8; ++j) row[lane + 32 * j] = v[j] * inv;
}

// softmax over P=9 within each (b,n,head)
__global__ void k_softmax9(float* __restrict__ a, long groups){
  long i = (long)blockIdx.x * 256 + threadIdx.x;
  if (i >= groups) return;
  long row = i / NHEAD; int h = (int)(i % NHEAD);
  float* p = a + row * (NHEAD * NPTS) + h * NPTS;
  float mx = -1e30f;
  for (int j = 0; j < NPTS; ++j) mx = fmaxf(mx, p[j]);
  float e[NPTS]; float s = 0.f;
  for (int j = 0; j < NPTS; ++j){ e[j] = __expf(p[j] - mx); s += e[j]; }
  float inv = 1.0f / s;
  for (int j = 0; j < NPTS; ++j) p[j] = e[j] * inv;
}

// bilinear align_corners upsample 16x16 -> 64x64 of [B,nN,C]-layout, accumulate
__global__ void k_upsample_add(const float* __restrict__ o, float* __restrict__ accum, int B){
  long total = (long)B * NTOK * CDIM;
  long i = (long)blockIdx.x * 256 + threadIdx.x;
  if (i >= total) return;
  int c = (int)(i % CDIM); long r = i / CDIM;
  int n = (int)(r % NTOK); int b = (int)(r / NTOK);
  int y = n >> 6, x = n & 63;
  float sy = y * 15.0f / 63.0f, sx = x * 15.0f / 63.0f;
  int iy = (int)sy; if (iy > 14) iy = 14;
  int ix = (int)sx; if (ix > 14) ix = 14;
  float wy = sy - iy, wx = sx - ix;
  const float* base = o + (long)b * NN * CDIM + c;
  float v00 = base[(long)(iy * 16 + ix) * CDIM];
  float v01 = base[(long)(iy * 16 + ix + 1) * CDIM];
  float v10 = base[(long)((iy + 1) * 16 + ix) * CDIM];
  float v11 = base[(long)((iy + 1) * 16 + ix + 1) * CDIM];
  accum[i] += (1 - wy) * ((1 - wx) * v00 + wx * v01) + wy * ((1 - wx) * v10 + wx * v11);
}

// MSDeformAttn sampling: one wave per (b,n,head), lane = channel d
__global__ void k_deform_sample(const float* __restrict__ value, const float* __restrict__ off,
                                const float* __restrict__ aw, float* __restrict__ out, int B){
  long wid = ((long)blockIdx.x * 256 + threadIdx.x) >> 5;
  int lane = threadIdx.x & 31;
  long total = (long)B * NTOK * NHEAD;
  if (wid >= total) return;
  int h = (int)(wid % NHEAD); long bn = wid / NHEAD;
  int n = (int)(bn % NTOK); int b = (int)(bn / NTOK);
  int y = n >> 6, x = n & 63;
  float rx = (x + 0.5f) / 64.0f, ry = (y + 0.5f) / 64.0f;
  const float* offp = off + ((long)b * NTOK + n) * (NHEAD * NPTS * 2) + h * (NPTS * 2);
  const float* awp  = aw  + ((long)b * NTOK + n) * (NHEAD * NPTS)     + h * NPTS;
  const float* vb   = value + (long)b * NTOK * CDIM + h * DHEAD + lane;
  float acc = 0.f;
#pragma unroll
  for (int p = 0; p < NPTS; ++p){
    float ox = offp[p * 2 + 0], oy = offp[p * 2 + 1];
    float px = (rx + ox / 64.0f) * 64.0f - 0.5f;
    float py = (ry + oy / 64.0f) * 64.0f - 0.5f;
    float fx = floorf(px), fy = floorf(py);
    float lx = px - fx, ly = py - fy;
    int x0 = (int)fx, y0 = (int)fy;
    float wgt = awp[p];
    float s = 0.f;
#pragma unroll
    for (int dy = 0; dy < 2; ++dy)
#pragma unroll
      for (int dx = 0; dx < 2; ++dx){
        int xc = x0 + dx, yc = y0 + dy;
        bool valid = (xc >= 0) && (xc < 64) && (yc >= 0) && (yc < 64);
        int cx = xc < 0 ? 0 : (xc > 63 ? 63 : xc);
        int cy = yc < 0 ? 0 : (yc > 63 ? 63 : yc);
        float wcorn = (dx ? lx : 1.0f - lx) * (dy ? ly : 1.0f - ly);
        if (valid) s += wcorn * vb[(long)(cy * 64 + cx) * CDIM];
      }
    acc += wgt * s;
  }
  out[((long)b * NTOK + n) * CDIM + h * DHEAD + lane] = acc;
}

// ---------------- host orchestration ----------------
static inline dim3 ew(long n){ return dim3((unsigned)((n + 255) / 256)); }

extern "C" void kernel_launch(void* const* d_in, const int* in_sizes, int n_in,
                              void* d_out, int out_size, void* d_ws, size_t ws_size,
                              hipStream_t stream)
{
  (void)n_in; (void)out_size; (void)ws_size;
  const float* x       = (const float*)d_in[0];
  /* d_in[1] = prior_mask, unused by the reference math */
  const float* s_x     = (const float*)d_in[2];
  const float* p_lvl   = (const float*)d_in[3];
  const float* p_srkvw = (const float*)d_in[4];
  const float* p_srkvb = (const float*)d_in[5];
  const float* p_nkvg  = (const float*)d_in[6];
  const float* p_nkvb  = (const float*)d_in[7];
  const float* p_srw   = (const float*)d_in[8];
  const float* p_srb   = (const float*)d_in[9];
  const float* p_ng    = (const float*)d_in[10];
  const float* p_nb    = (const float*)d_in[11];
  const float* p_qw    = (const float*)d_in[12];
  const float* p_kvw   = (const float*)d_in[13];
  const float* p_projw = (const float*)d_in[14];
  const float* p_projb = (const float*)d_in[15];
  const float* p_fc1   = (const float*)d_in[16];
  const float* p_fc2   = (const float*)d_in[17];
  const float* p_lncag = (const float*)d_in[18];
  const float* p_lncab = (const float*)d_in[19];
  const float* p_lncfg = (const float*)d_in[20];
  const float* p_lncfb = (const float*)d_in[21];
  const float* p_offw  = (const float*)d_in[22];
  const float* p_offb  = (const float*)d_in[23];
  const float* p_aww   = (const float*)d_in[24];
  const float* p_awb   = (const float*)d_in[25];
  const float* p_valw  = (const float*)d_in[26];
  const float* p_valb  = (const float*)d_in[27];
  const float* p_outw  = (const float*)d_in[28];
  const float* p_outb  = (const float*)d_in[29];
  const float* p_fs1   = (const float*)d_in[30];
  const float* p_fs2   = (const float*)d_in[31];
  const float* p_lnsag = (const float*)d_in[32];
  const float* p_lnsab = (const float*)d_in[33];
  const float* p_lnsfg = (const float*)d_in[34];
  const float* p_lnsfb = (const float*)d_in[35];

  int B = in_sizes[0] / (CDIM * NTOK);
  int S = in_sizes[2] / (B * NTOK * CDIM);
  long BNC = (long)B * NTOK * CDIM;
  long BnC = (long)B * NN * CDIM;

  float* w = (float*)d_ws;
  size_t offf = 0;
  auto alloc = [&](long nf){ float* p = w + offf; offf += (size_t)nf; return p; };
  float* pos    = alloc((long)NTOK * CDIM);
  float* q      = alloc(BNC);
  float* t0     = alloc(BNC);
  float* t1     = alloc(BNC);
  float* t2     = alloc(BNC);
  float* hid    = alloc(BNC * 3);
  float* Abuf   = alloc((long)B * NN * KCONV);
  float* nqbuf  = alloc(BnC);
  float* xbuf   = alloc(BnC);
  float* qhbuf  = alloc(BnC);
  float* obuf   = alloc(BnC);
  float* oproj  = alloc(BnC);
  float* kvbuf  = alloc(BnC * 2);
  float* Sbuf   = alloc((long)B * NHEAD * NN * NN);
  float* offbuf = alloc((long)B * NTOK * NHEAD * NPTS * 2);
  float* awbuf  = alloc((long)B * NTOK * NHEAD * NPTS);

  auto gemm = [&](const float* A, const float* Bm, const float* bias, float* Cm,
                  int M, int N, int K, long lda, long ldb, long ldc,
                  int batches, int innerB,
                  long sAo, long sAi, long sBo, long sBi, long sCo, long sCi,
                  bool transB, float alpha, bool relu){
    dim3 grid((N + 127) / 128, (M + 127) / 128, batches);
    if (transB)
      gemm_wmma_f16<1><<<grid, 256, 0, stream>>>(A, Bm, bias, Cm, M, N, K, lda, ldb, ldc,
                                                 sAo, sAi, sBo, sBi, sCo, sCi,
                                                 innerB, alpha, relu ? 1 : 0);
    else
      gemm_wmma_f16<0><<<grid, 256, 0, stream>>>(A, Bm, bias, Cm, M, N, K, lda, ldb, ldc,
                                                 sAo, sAi, sBo, sBi, sCo, sCi,
                                                 innerB, alpha, relu ? 1 : 0);
  };

  long rowsBN = (long)B * NTOK;
  long rowsBn = (long)B * NN;
  const float scale = 0.17677669529663687f;   // 32^-0.5

  k_pos<<<ew((long)NTOK * CDIM), 256, 0, stream>>>(p_lvl, pos);
  k_transpose_x<<<ew(BNC), 256, 0, stream>>>(x, q, B);

  // ---------------- cross-attention blocks ----------------
  for (int l = 0; l < NLAYER; ++l){
    const float* srkv_w = p_srkvw + (long)l * CDIM * KCONV;
    const float* sr_w   = p_srw   + (long)l * CDIM * KCONV;

    // K/V from spatially-reduced q features
    k_im2col<<<ew((long)B * NN * KCONV), 256, 0, stream>>>(q, (long)NTOK * CDIM, Abuf, B);
    gemm(Abuf, srkv_w, p_srkvb + l * CDIM, nqbuf, B * NN, CDIM, KCONV,
         KCONV, KCONV, CDIM, 1, 1, 0,0,0,0,0,0, true, 1.0f, false);
    k_layernorm<<<ew(rowsBn * 32), 256, 0, stream>>>(nqbuf, p_nkvg + l * CDIM, p_nkvb + l * CDIM, rowsBn);
    gemm(nqbuf, p_kvw + (long)l * CDIM * 2 * CDIM, nullptr, kvbuf, B * NN, 2 * CDIM, CDIM,
         CDIM, 2 * CDIM, 2 * CDIM, 1, 1, 0,0,0,0,0,0, false, 1.0f, false);

    k_fill_zero<<<ew(BNC), 256, 0, stream>>>(t0, BNC);
    for (int st = 0; st < S; ++st){
      k_im2col<<<ew((long)B * NN * KCONV), 256, 0, stream>>>(
          s_x + (long)st * NTOK * CDIM, (long)S * NTOK * CDIM, Abuf, B);
      gemm(Abuf, sr_w, p_srb + l * CDIM, xbuf, B * NN, CDIM, KCONV,
           KCONV, KCONV, CDIM, 1, 1, 0,0,0,0,0,0, true, 1.0f, false);
      k_layernorm<<<ew(rowsBn * 32), 256, 0, stream>>>(xbuf, p_ng + l * CDIM, p_nb + l * CDIM, rowsBn);
      gemm(xbuf, p_qw + (long)l * CDIM * CDIM, nullptr, qhbuf, B * NN, CDIM, CDIM,
           CDIM, CDIM, CDIM, 1, 1, 0,0,0,0,0,0, false, 1.0f, false);
      // S = Qh * K^T * scale, batched over (b, head)
      gemm(qhbuf, kvbuf, nullptr, Sbuf, NN, NN, DHEAD,
           CDIM, 2 * CDIM, NN,
           B * NHEAD, NHEAD,
           (long)NN * CDIM, DHEAD,
           (long)NN * 2 * CDIM, DHEAD,
           (long)NHEAD * NN * NN, (long)NN * NN,
           true, scale, false);
      k_softmax256<<<ew((long)B * NHEAD * NN * 32), 256, 0, stream>>>(Sbuf, (long)B * NHEAD * NN);
      // O = S * V  (written back as [B,nN,C])
      gemm(Sbuf, kvbuf + CDIM, nullptr, obuf, NN, DHEAD, NN,
           NN, 2 * CDIM, CDIM,
           B * NHEAD, NHEAD,
           (long)NHEAD * NN * NN, (long)NN * NN,
           (long)NN * 2 * CDIM, DHEAD,
           (long)NN * CDIM, DHEAD,
           false, 1.0f, false);
      gemm(obuf, p_projw + (long)l * CDIM * CDIM, p_projb + l * CDIM, oproj,
           B * NN, CDIM, CDIM, CDIM, CDIM, CDIM, 1, 1, 0,0,0,0,0,0, false, 1.0f, false);
      k_upsample_add<<<ew(BNC), 256, 0, stream>>>(oproj, t0, B);
    }
    k_res_layernorm<<<ew(rowsBN * 32), 256, 0, stream>>>(
        q, q, t0, 1.0f / (float)S, p_lncag + l * CDIM, p_lncab + l * CDIM, rowsBN);

    // FFN
    gemm(q, p_fc1 + (long)l * CDIM * 3 * CDIM, nullptr, hid, B * NTOK, 3 * CDIM, CDIM,
         CDIM, 3 * CDIM, 3 * CDIM, 1, 1, 0,0,0,0,0,0, false, 1.0f, true);
    gemm(hid, p_fc2 + (long)l * 3 * CDIM * CDIM, nullptr, t1, B * NTOK, CDIM, 3 * CDIM,
         3 * CDIM, CDIM, CDIM, 1, 1, 0,0,0,0,0,0, false, 1.0f, false);
    k_res_layernorm<<<ew(rowsBN * 32), 256, 0, stream>>>(
        q, q, t1, 1.0f, p_lncfg + l * CDIM, p_lncfb + l * CDIM, rowsBN);
  }

  // ---------------- deformable self-attention blocks ----------------
  for (int l = 0; l < NLAYER; ++l){
    k_add_pos<<<ew(BNC), 256, 0, stream>>>(q, pos, t1, B);     // qpos
    gemm(q, p_valw + (long)l * CDIM * CDIM, p_valb + l * CDIM, t2, B * NTOK, CDIM, CDIM,
         CDIM, CDIM, CDIM, 1, 1, 0,0,0,0,0,0, false, 1.0f, false);
    gemm(t1, p_offw + (long)l * CDIM * (NHEAD * NPTS * 2), p_offb + l * (NHEAD * NPTS * 2),
         offbuf, B * NTOK, NHEAD * NPTS * 2, CDIM,
         CDIM, NHEAD * NPTS * 2, NHEAD * NPTS * 2, 1, 1, 0,0,0,0,0,0, false, 1.0f, false);
    gemm(t1, p_aww + (long)l * CDIM * (NHEAD * NPTS), p_awb + l * (NHEAD * NPTS),
         awbuf, B * NTOK, NHEAD * NPTS, CDIM,
         CDIM, NHEAD * NPTS, NHEAD * NPTS, 1, 1, 0,0,0,0,0,0, false, 1.0f, false);
    k_softmax9<<<ew((long)B * NTOK * NHEAD), 256, 0, stream>>>(awbuf, (long)B * NTOK * NHEAD);
    k_deform_sample<<<ew((long)B * NTOK * NHEAD * 32), 256, 0, stream>>>(t2, offbuf, awbuf, t0, B);
    gemm(t0, p_outw + (long)l * CDIM * CDIM, p_outb + l * CDIM, t1, B * NTOK, CDIM, CDIM,
         CDIM, CDIM, CDIM, 1, 1, 0,0,0,0,0,0, false, 1.0f, false);
    k_res_layernorm<<<ew(rowsBN * 32), 256, 0, stream>>>(
        q, q, t1, 1.0f, p_lnsag + l * CDIM, p_lnsab + l * CDIM, rowsBN);

    gemm(q, p_fs1 + (long)l * CDIM * 3 * CDIM, nullptr, hid, B * NTOK, 3 * CDIM, CDIM,
         CDIM, 3 * CDIM, 3 * CDIM, 1, 1, 0,0,0,0,0,0, false, 1.0f, true);
    gemm(hid, p_fs2 + (long)l * 3 * CDIM * CDIM, nullptr, t1, B * NTOK, CDIM, 3 * CDIM,
         3 * CDIM, CDIM, CDIM, 1, 1, 0,0,0,0,0,0, false, 1.0f, false);
    float* dst = (l == NLAYER - 1) ? (float*)d_out : q;
    k_res_layernorm<<<ew(rowsBN * 32), 256, 0, stream>>>(
        dst, q, t1, 1.0f, p_lnsfg + l * CDIM, p_lnsfb + l * CDIM, rowsBN);
  }
}
